// AngularPenaltySMLoss_71107478552756
// MI455X (gfx1250) — matrix operand order repaired and checked by reading, not compiled, not run on hardware
//
#include <hip/hip_runtime.h>
#include <hip/hip_bf16.h>

// ---------------------------------------------------------------------------
// ArcFace loss, fused for MI455X (gfx1250, wave32, WMMA).
//   K1: normalize rows of x -> bf16 hi/lo split in workspace, zero row sums
//   K2: exact fp32 target logits (label gather)
//   K3: fused  exp-sum( xn @ W^T * 64 )  via bf16x3 v_wmma_f32_16x16x32_bf16,
//       W converted fp32->bf16(hi/lo) once per block into 266KB dynamic LDS
//       so W streams from HBM exactly once (~205MB @ 23.3TB/s).
//   K4: margin math + mean reduction -> scalar loss
// ---------------------------------------------------------------------------

typedef __attribute__((ext_vector_type(16))) __bf16        v16bf;
typedef __attribute__((ext_vector_type(8)))  float         v8f;
typedef __attribute__((ext_vector_type(4)))  float         v4f;
typedef __attribute__((ext_vector_type(4)))  unsigned int  v4u;

union Frag16 { v16bf bf; v4u u4[2]; };

static constexpr int BB    = 1024;     // batch
static constexpr int CC    = 100000;   // classes
static constexpr int DD    = 512;      // feature dim
static constexpr int CTILE = 128;      // classes per block
static constexpr int LDSSTR = DD + 8;  // padded bf16 row stride (bank-conflict free)
static constexpr float S_SCALE = 64.0f;
static constexpr float COS_M   = 0.8775825618903728f;  // cos(0.5)
static constexpr float SIN_M   = 0.4794255386042030f;  // sin(0.5)

__device__ __forceinline__ unsigned short f2bf(float f) {
  unsigned u = __float_as_uint(f);
  unsigned r = u + 0x7FFFu + ((u >> 16) & 1u);   // RNE truncation
  return (unsigned short)(r >> 16);
}
__device__ __forceinline__ float bf2f(unsigned short h) {
  return __uint_as_float(((unsigned)h) << 16);
}

// ---------------------------------------------------------------------------
// K1: xn = x / max(||x||,1e-12); split into bf16 hi/lo; init row_sum = 0
// ---------------------------------------------------------------------------
__global__ __launch_bounds__(256)
void k_normalize(const float* __restrict__ x,
                 unsigned short* __restrict__ xnh,
                 unsigned short* __restrict__ xnl,
                 float* __restrict__ inv_norm,
                 float* __restrict__ row_sum) {
  const int b = blockIdx.x, tid = threadIdx.x;
  const size_t base = (size_t)b * DD;
  float v0 = x[base + tid];
  float v1 = x[base + tid + 256];
  float ss = v0 * v0 + v1 * v1;
#pragma unroll
  for (int m = 1; m < 32; m <<= 1) ss += __shfl_xor(ss, m, 32);
  __shared__ float ps[8];
  if ((tid & 31) == 0) ps[tid >> 5] = ss;
  __syncthreads();
  float tot = ps[0] + ps[1] + ps[2] + ps[3] + ps[4] + ps[5] + ps[6] + ps[7];
  float inv = 1.0f / fmaxf(sqrtf(tot), 1e-12f);
  float n0 = v0 * inv, n1 = v1 * inv;
  unsigned short h0 = f2bf(n0), h1 = f2bf(n1);
  xnh[base + tid]        = h0;
  xnl[base + tid]        = f2bf(n0 - bf2f(h0));
  xnh[base + tid + 256]  = h1;
  xnl[base + tid + 256]  = f2bf(n1 - bf2f(h1));
  if (tid == 0) { inv_norm[b] = inv; row_sum[b] = 0.0f; }
}

// ---------------------------------------------------------------------------
// K2: target[b] = (x[b] . W[label[b]]) * inv_norm[b]   (exact fp32)
// ---------------------------------------------------------------------------
__global__ __launch_bounds__(256)
void k_target(const float* __restrict__ x,
              const int* __restrict__ labels,
              const float* __restrict__ W,
              const float* __restrict__ inv_norm,
              float* __restrict__ target) {
  const int b = blockIdx.x, tid = threadIdx.x;
  const size_t xb = (size_t)b * DD;
  const size_t wb = (size_t)labels[b] * DD;
  float d = x[xb + tid] * W[wb + tid] + x[xb + tid + 256] * W[wb + tid + 256];
#pragma unroll
  for (int m = 1; m < 32; m <<= 1) d += __shfl_xor(d, m, 32);
  __shared__ float ps[8];
  if ((tid & 31) == 0) ps[tid >> 5] = d;
  __syncthreads();
  if (tid == 0) {
    float tot = ps[0] + ps[1] + ps[2] + ps[3] + ps[4] + ps[5] + ps[6] + ps[7];
    target[b] = tot * inv_norm[b];
  }
}

// ---------------------------------------------------------------------------
// K3: per 128-class tile: row_sum[b] += sum_c exp(64 * xn[b].W[c])
//     bf16x3 WMMA GEMM, W tile held in LDS as bf16 hi/lo.
// Block: 256 thr = 8 waves as 2(M) x 4(N); wave tile 32x32 (2x2 frags).
// ---------------------------------------------------------------------------
__global__ __launch_bounds__(256)
void k_gemm_expsum(const unsigned short* __restrict__ xnh,
                   const unsigned short* __restrict__ xnl,
                   const float* __restrict__ W,
                   float* __restrict__ row_sum) {
  extern __shared__ char smem[];
  unsigned short* Whi = (unsigned short*)smem;
  unsigned short* Wlo = Whi + CTILE * LDSSTR;
  float* rsum = (float*)(Wlo + CTILE * LDSSTR);

  const int tid  = threadIdx.x;
  const int lane = tid & 31;
  const int wave = tid >> 5;
  const int c0   = blockIdx.x * CTILE;

  // ---- stage W tile: fp32 -> bf16 hi/lo into LDS (done once per block) ----
  for (int i = tid; i < CTILE * (DD / 4); i += 256) {
    const int row  = i >> 7;            // DD/4 == 128
    const int col4 = (i & 127) << 2;
    const int c = c0 + row;
    v4f w = {};
    if (c < CC) w = *(const v4f*)(W + (size_t)c * DD + col4);
    union { unsigned short h[4]; unsigned long long u; } ph, pl;
#pragma unroll
    for (int j = 0; j < 4; ++j) {
      unsigned short h = f2bf(w[j]);
      ph.h[j] = h;
      pl.h[j] = f2bf(w[j] - bf2f(h));
    }
    *(unsigned long long*)(Whi + row * LDSSTR + col4) = ph.u;
    *(unsigned long long*)(Wlo + row * LDSSTR + col4) = pl.u;
  }
  if (tid < 64) rsum[tid] = 0.0f;
  __syncthreads();

  const int wm  = wave >> 2;       // 0..1 : M position
  const int wn  = wave & 3;        // 0..3 : N position
  const int l15 = lane & 15;
  const int lh  = lane >> 4;
  const v8f zacc = {0.f, 0.f, 0.f, 0.f, 0.f, 0.f, 0.f, 0.f};

  for (int chunk = 0; chunk < BB / 64; ++chunk) {
    const int m0 = chunk * 64 + wm * 32;
    v8f acc[2][2];
    acc[0][0] = zacc; acc[0][1] = zacc; acc[1][0] = zacc; acc[1][1] = zacc;

    for (int k0 = 0; k0 < DD; k0 += 32) {
      // A frags (ISA 16-bit A 16x32 layout): lane m; lh=0 -> K {0..7,16..23}
      Frag16 ah[2], al[2];
#pragma unroll
      for (int mi = 0; mi < 2; ++mi) {
        const size_t off = (size_t)(m0 + mi * 16 + l15) * DD + k0 + lh * 8;
        ah[mi].u4[0] = *(const v4u*)(xnh + off);
        ah[mi].u4[1] = *(const v4u*)(xnh + off + 16);
        al[mi].u4[0] = *(const v4u*)(xnl + off);
        al[mi].u4[1] = *(const v4u*)(xnl + off + 16);
      }
      // B frags (ISA B KxN layout): lane n = l15; lh=0 -> K {0..7,16..23}
      Frag16 bh[2], bl[2];
#pragma unroll
      for (int ni = 0; ni < 2; ++ni) {
        const int boff = (wn * 32 + ni * 16 + l15) * LDSSTR + k0 + lh * 8;
        bh[ni].u4[0] = *(const v4u*)(Whi + boff);
        bh[ni].u4[1] = *(const v4u*)(Whi + boff + 16);
        bl[ni].u4[0] = *(const v4u*)(Wlo + boff);
        bl[ni].u4[1] = *(const v4u*)(Wlo + boff + 16);
      }
#pragma unroll
      for (int mi = 0; mi < 2; ++mi)
#pragma unroll
        for (int ni = 0; ni < 2; ++ni) {
          acc[mi][ni] = __builtin_amdgcn_wmma_f32_16x16x32_bf16(
              false, ah[mi].bf, false, bh[ni].bf, (short)0, acc[mi][ni], false, false);
          acc[mi][ni] = __builtin_amdgcn_wmma_f32_16x16x32_bf16(
              false, al[mi].bf, false, bh[ni].bf, (short)0, acc[mi][ni], false, false);
          acc[mi][ni] = __builtin_amdgcn_wmma_f32_16x16x32_bf16(
              false, ah[mi].bf, false, bl[ni].bf, (short)0, acc[mi][ni], false, false);
        }
    }

    // epilogue: e = exp(64*logit), mask padded classes, reduce over N
#pragma unroll
    for (int mi = 0; mi < 2; ++mi) {
#pragma unroll
      for (int v = 0; v < 8; ++v) {
        float e = 0.0f;
#pragma unroll
        for (int ni = 0; ni < 2; ++ni) {
          const int cglob = c0 + wn * 32 + ni * 16 + l15;
          const float ex = __expf(S_SCALE * acc[mi][ni][v]);
          e += (cglob < CC) ? ex : 0.0f;
        }
#pragma unroll
        for (int m = 1; m < 16; m <<= 1) e += __shfl_xor(e, m, 32);
        if (l15 == 0)
          atomicAdd(&rsum[wm * 32 + mi * 16 + v + 8 * lh], e);  // ds_add_f32
      }
    }
    __syncthreads();
    if (tid < 64) {
      atomicAdd(&row_sum[chunk * 64 + tid], rsum[tid]);  // global_atomic_add_f32
      rsum[tid] = 0.0f;
    }
    __syncthreads();
  }
}

// ---------------------------------------------------------------------------
// K4: loss = -mean( num - log(exp(num) + (row_sum - exp(s*t))) )
//     num  = s * (t*cos m - sqrt(1-t^2)*sin m)   (== s*cos(acos(t)+m))
// ---------------------------------------------------------------------------
__global__ __launch_bounds__(1024)
void k_finalize(const float* __restrict__ target,
                const float* __restrict__ row_sum,
                float* __restrict__ out) {
  const int tid = threadIdx.x;
  float t = target[tid];
  t = fminf(fmaxf(t, -1.0f + 1e-7f), 1.0f - 1e-7f);
  const float num  = S_SCALE * (t * COS_M - sqrtf(fmaxf(1.0f - t * t, 0.0f)) * SIN_M);
  const float excl = row_sum[tid] - __expf(S_SCALE * t);
  float L = num - __logf(__expf(num) + excl);
#pragma unroll
  for (int m = 1; m < 32; m <<= 1) L += __shfl_xor(L, m, 32);
  __shared__ float ps[32];
  if ((tid & 31) == 0) ps[tid >> 5] = L;
  __syncthreads();
  if (tid < 32) {
    float s = ps[tid];
#pragma unroll
    for (int m = 1; m < 32; m <<= 1) s += __shfl_xor(s, m, 32);
    if (tid == 0) out[0] = -(s * (1.0f / (float)BB));
  }
}

// ---------------------------------------------------------------------------
extern "C" void kernel_launch(void* const* d_in, const int* in_sizes, int n_in,
                              void* d_out, int out_size, void* d_ws, size_t ws_size,
                              hipStream_t stream) {
  (void)in_sizes; (void)n_in; (void)out_size; (void)ws_size;
  const float* x      = (const float*)d_in[0];
  const int*   labels = (const int*)d_in[1];
  const float* W      = (const float*)d_in[2];
  float*       out    = (float*)d_out;

  // workspace layout (~2.1 MB total)
  char* ws = (char*)d_ws;
  unsigned short* xnh = (unsigned short*)ws;                       // 1 MB
  unsigned short* xnl = (unsigned short*)(ws + (size_t)BB * DD * 2);      // 1 MB
  float* inv_norm = (float*)(ws + (size_t)BB * DD * 4);            // 4 KB
  float* row_sum  = (float*)(ws + (size_t)BB * DD * 4 + BB * 4);   // 4 KB
  float* target   = (float*)(ws + (size_t)BB * DD * 4 + BB * 8);   // 4 KB

  const int    nblocks = (CC + CTILE - 1) / CTILE;                 // 782
  const size_t smem    = (size_t)2 * CTILE * LDSSTR * 2 + 64 * 4;  // 266,496 B

  (void)hipFuncSetAttribute(reinterpret_cast<const void*>(k_gemm_expsum),
                            hipFuncAttributeMaxDynamicSharedMemorySize, (int)smem);

  k_normalize<<<BB, 256, 0, stream>>>(x, xnh, xnl, inv_norm, row_sum);
  k_target<<<BB, 256, 0, stream>>>(x, labels, W, inv_norm, target);
  k_gemm_expsum<<<nblocks, 256, smem, stream>>>(xnh, xnl, W, row_sum);
  k_finalize<<<1, 1024, 0, stream>>>(target, row_sum, out);
}
